// Critic_88450556493905
// MI455X (gfx1250) — compile-verified, matrix operand order checked by reading, not run on hardware
//
#include <hip/hip_runtime.h>

typedef __attribute__((ext_vector_type(16))) _Float16 v16h;
typedef __attribute__((ext_vector_type(8)))  _Float16 v8h;
typedef __attribute__((ext_vector_type(4)))  _Float16 v4h;
typedef __attribute__((ext_vector_type(8)))  float    v8f;
typedef __attribute__((ext_vector_type(4)))  float    v4f;

#define THRESH   0.5f
#define T_STEPS  16
#define Ddim     256
#define Hdim     512
#define MTILE    32
#define NTHREADS 256

// Build a 16-half A/B fragment per the CDNA5 16-bit operand layout:
// lanes 0-15 hold K in [k..k+7] and [k+16..k+23]; lanes 16-31 hold
// [k+8..k+15] and [k+24..k+31] (k0/k1 already include the half-wave shift).
__device__ __forceinline__ v16h make_frag(const _Float16* __restrict__ rowPtr,
                                          int k0, int k1) {
    v8h p = *(const v8h*)(rowPtr + k0);
    v8h q = *(const v8h*)(rowPtr + k1);
    return __builtin_shufflevector(p, q, 0,1,2,3,4,5,6,7,8,9,10,11,12,13,14,15);
}

// ---- prep: W1 [256x512] f32 -> W1^T [512x256] f16 in workspace
__global__ void prep_w1t(const float* __restrict__ W1, _Float16* __restrict__ w1t) {
    const int n = blockIdx.x;          // 0..511 (output row = column of W1)
    const int k = threadIdx.x;         // 0..255
    w1t[(size_t)n * Ddim + k] = (_Float16)W1[(size_t)k * Hdim + n];
}

// ---- prep: W2 [512x512] f32 -> W2^T [512x512] f16 in workspace
__global__ void prep_w2t(const float* __restrict__ W2, _Float16* __restrict__ w2t) {
    const int n = blockIdx.x;          // 0..511
    for (int k = threadIdx.x; k < Hdim; k += NTHREADS)
        w2t[(size_t)n * Hdim + k] = (_Float16)W2[(size_t)k * Hdim + n];
}

// ---- fused SNN critic: GEMM1 + 16-step LIF recurrence entirely on-chip
__global__ __launch_bounds__(NTHREADS, 1)
void snn_critic_fused(const float* __restrict__ X,
                      const float* __restrict__ b1g,
                      const float* __restrict__ b2g,
                      const float* __restrict__ W3g,
                      const float* __restrict__ b3g,
                      const _Float16* __restrict__ w1t,
                      const _Float16* __restrict__ w2t,
                      float* __restrict__ out)
{
    __shared__ __align__(16) float    s_dv1 [MTILE][Hdim];      // 64 KB
    __shared__ __align__(16) float    s_mem1[MTILE][Hdim];      // 64 KB
    __shared__ __align__(16) float    s_mem2[MTILE][Hdim];      // 64 KB
    __shared__ __align__(16) _Float16 s_spk [MTILE][Hdim + 8];  // 32.5 KB (s1 then s2)
    __shared__ __align__(16) _Float16 s_x   [MTILE][Ddim + 8];  // 16.5 KB
    __shared__ __align__(16) float s_w3[Hdim];
    __shared__ __align__(16) float s_b2[Hdim];
    __shared__ float s_part[MTILE][8];
    __shared__ float s_vout[MTILE];

    const int tid     = threadIdx.x;
    const int wave    = tid >> 5;
    const int lane    = tid & 31;
    const int half    = lane >> 4;
    const int lrow    = lane & 15;
    const int rowBase = blockIdx.x * MTILE;

    // ---------- init: stage X tile as f16 (vectorized), zero LIF state
    for (int i = tid * 4; i < MTILE * Ddim; i += NTHREADS * 4) {
        const int r = i >> 8, c = i & (Ddim - 1);
        v4f x = *(const v4f*)&X[(size_t)(rowBase + r) * Ddim + c];
        v4h h;
        #pragma unroll
        for (int e = 0; e < 4; ++e) h[e] = (_Float16)x[e];
        *(v4h*)&s_x[r][c] = h;
    }
    {
        const v4f z = {};
        for (int i = tid * 4; i < MTILE * Hdim; i += NTHREADS * 4) {
            const int r = i >> 9, c = i & (Hdim - 1);
            *(v4f*)&s_mem1[r][c] = z;
            *(v4f*)&s_mem2[r][c] = z;
        }
    }
    for (int i = tid; i < Hdim; i += NTHREADS) { s_w3[i] = W3g[i]; s_b2[i] = b2g[i]; }
    if (tid < MTILE) s_vout[tid] = 0.0f;
    // warm L2 with W2^T (hint only): global_prefetch_b8
    __builtin_prefetch(w2t + (size_t)tid * 1024, 0, 1);
    __syncthreads();

    // ---------- GEMM1: dv1 = X @ W1 + b1 (A: LDS f16, B: global W1^T f16)
    {
        v8f acc[2][4] = {};
        #pragma unroll
        for (int kk = 0; kk < Ddim / 32; ++kk) {
            const int k  = kk * 32;
            const int k0 = k + half * 8;
            const int k1 = k + 16 + half * 8;
            v16h a0 = make_frag(&s_x[lrow][0],      k0, k1);
            v16h a1 = make_frag(&s_x[16 + lrow][0], k0, k1);
            #pragma unroll
            for (int j = 0; j < 4; ++j) {
                const int n = (wave * 4 + j) * 16 + lrow;
                v16h b = make_frag(w1t + (size_t)n * Ddim, k0, k1);
                acc[0][j] = __builtin_amdgcn_wmma_f32_16x16x32_f16(
                    false, a0, false, b, (short)0, acc[0][j], false, false);
                acc[1][j] = __builtin_amdgcn_wmma_f32_16x16x32_f16(
                    false, a1, false, b, (short)0, acc[1][j], false, false);
            }
        }
        // epilogue: C layout -> LDS dv1 (+bias). VGPR i: row = i + 8*half.
        #pragma unroll
        for (int j = 0; j < 4; ++j) {
            const int col  = (wave * 4 + j) * 16 + lrow;
            const float bv = b1g[col];
            #pragma unroll
            for (int rt = 0; rt < 2; ++rt)
                #pragma unroll
                for (int i = 0; i < 8; ++i)
                    s_dv1[rt * 16 + half * 8 + i][col] = acc[rt][j][i] + bv;
        }
    }
    __syncthreads();

    const float b3v = b3g[0];

    // ---------- 16-step LIF recurrence, fully on-chip
    for (int t = 0; t < T_STEPS; ++t) {
        // LIF layer 1 (tau=2): mem=(mem+dv1)/2, spike>0.5, hard reset.
        // Vectorized: 8 elems/iter -> b128 LDS transactions.
        for (int i = tid * 8; i < MTILE * Hdim; i += NTHREADS * 8) {
            const int r = i >> 9, c = i & (Hdim - 1);
            v4f m0 = (*(const v4f*)&s_mem1[r][c]     + *(const v4f*)&s_dv1[r][c])     * 0.5f;
            v4f m1 = (*(const v4f*)&s_mem1[r][c + 4] + *(const v4f*)&s_dv1[r][c + 4]) * 0.5f;
            v8h sp;
            v4f w0, w1;
            #pragma unroll
            for (int e = 0; e < 4; ++e) {
                const float a = m0[e];
                const float sa = (a > THRESH) ? 1.0f : 0.0f;
                w0[e] = a * (1.0f - sa);
                sp[e] = (_Float16)sa;
                const float b = m1[e];
                const float sb = (b > THRESH) ? 1.0f : 0.0f;
                w1[e] = b * (1.0f - sb);
                sp[4 + e] = (_Float16)sb;
            }
            *(v4f*)&s_mem1[r][c]     = w0;
            *(v4f*)&s_mem1[r][c + 4] = w1;
            *(v8h*)&s_spk[r][c]      = sp;
        }
        __syncthreads();

        // GEMM2: dv2 = s1 @ W2 (A: spikes in LDS, B: global W2^T, L2-resident)
        v8f acc[2][4] = {};
        #pragma unroll 4
        for (int kk = 0; kk < Hdim / 32; ++kk) {
            const int k  = kk * 32;
            const int k0 = k + half * 8;
            const int k1 = k + 16 + half * 8;
            v16h a0 = make_frag(&s_spk[lrow][0],      k0, k1);
            v16h a1 = make_frag(&s_spk[16 + lrow][0], k0, k1);
            #pragma unroll
            for (int j = 0; j < 4; ++j) {
                const int n = (wave * 4 + j) * 16 + lrow;
                v16h b = make_frag(w2t + (size_t)n * Hdim, k0, k1);
                acc[0][j] = __builtin_amdgcn_wmma_f32_16x16x32_f16(
                    false, a0, false, b, (short)0, acc[0][j], false, false);
                acc[1][j] = __builtin_amdgcn_wmma_f32_16x16x32_f16(
                    false, a1, false, b, (short)0, acc[1][j], false, false);
            }
        }
        __syncthreads();   // all waves done reading s1 from s_spk

        // LIF layer 2 fused into WMMA epilogue; s2 overwrites s_spk
        #pragma unroll
        for (int j = 0; j < 4; ++j) {
            const int col  = (wave * 4 + j) * 16 + lrow;
            const float bv = s_b2[col];
            #pragma unroll
            for (int rt = 0; rt < 2; ++rt)
                #pragma unroll
                for (int i = 0; i < 8; ++i) {
                    const int row = rt * 16 + half * 8 + i;
                    const float m  = (s_mem2[row][col] + acc[rt][j][i] + bv) * 0.5f;
                    const float sp = (m > THRESH) ? 1.0f : 0.0f;
                    s_mem2[row][col] = m * (1.0f - sp);
                    s_spk[row][col]  = (_Float16)sp;
                }
        }
        __syncthreads();

        // output GEMV: o = s2 @ W3 + b3 ; vout = (vout + o)/2 (vectorized)
        {
            const int r = tid >> 3, seg = tid & 7;
            const _Float16* sp = &s_spk[r][seg * 64];
            const float*    wp = &s_w3[seg * 64];
            float sum = 0.0f;
            #pragma unroll
            for (int cc = 0; cc < 64; cc += 8) {
                v8h s  = *(const v8h*)(sp + cc);
                v4f w0 = *(const v4f*)(wp + cc);
                v4f w1 = *(const v4f*)(wp + cc + 4);
                #pragma unroll
                for (int e = 0; e < 4; ++e) {
                    sum += (float)s[e]     * w0[e];
                    sum += (float)s[4 + e] * w1[e];
                }
            }
            s_part[r][seg] = sum;
        }
        __syncthreads();
        if (tid < MTILE) {
            float o = b3v;
            #pragma unroll
            for (int s = 0; s < 8; ++s) o += s_part[tid][s];
            s_vout[tid] = (s_vout[tid] + o) * 0.5f;
        }
        __syncthreads();
    }

    if (tid < MTILE) out[rowBase + tid] = s_vout[tid];
}

extern "C" void kernel_launch(void* const* d_in, const int* in_sizes, int n_in,
                              void* d_out, int out_size, void* d_ws, size_t ws_size,
                              hipStream_t stream) {
    const float* X  = (const float*)d_in[0];
    // d_in[1] = actions (unused by the forward pass)
    const float* W1 = (const float*)d_in[2];
    const float* b1 = (const float*)d_in[3];
    const float* W2 = (const float*)d_in[4];
    const float* b2 = (const float*)d_in[5];
    const float* W3 = (const float*)d_in[6];
    const float* b3 = (const float*)d_in[7];
    float* out = (float*)d_out;

    const int Bn = in_sizes[0] / Ddim;          // 65536

    _Float16* w1t = (_Float16*)d_ws;            // 512*256 f16 = 256 KB
    _Float16* w2t = w1t + (size_t)Hdim * Ddim;  // 512*512 f16 = 512 KB

    prep_w1t<<<Hdim, Ddim,     0, stream>>>(W1, w1t);
    prep_w2t<<<Hdim, NTHREADS, 0, stream>>>(W2, w2t);
    snn_critic_fused<<<Bn / MTILE, NTHREADS, 0, stream>>>(
        X, b1, b2, W3, b3, w1t, w2t, out);
}
